// sLSTMCell_49787260895631
// MI455X (gfx1250) — matrix operand order repaired and checked by reading, not compiled, not run on hardware
//
#include <hip/hip_runtime.h>
#include <hip/hip_bf16.h>

// Problem constants from the reference
#define BB 8
#define SS 4096
#define DD 1024
#define HH 4
#define HDIM 256
#define NGATE 4
#define BSROWS (BB * SS)   // 32768
#define BD (BB * DD)       // 8192 state lanes
#define NC 64              // scan chunks
#define LC (SS / NC)       // 64 steps per chunk

typedef __bf16 bf16_t;
typedef bf16_t v16bf __attribute__((ext_vector_type(16)));
typedef float  v8f   __attribute__((ext_vector_type(8)));
typedef unsigned int v4u __attribute__((ext_vector_type(4)));
typedef int v8i __attribute__((ext_vector_type(8)));
typedef int v4i __attribute__((ext_vector_type(4)));

union Frag128x2 {   // 8 VGPRs = 16 bf16, loadable as two 16B chunks
    v16bf v;
    uint4 u[2];
};

// ---------------------------------------------------------------------------
// Stage 0: fp32 -> bf16 conversion (x and weights) into workspace
// ---------------------------------------------------------------------------
__global__ void __launch_bounds__(256)
cvt_f32_to_bf16(const float* __restrict__ src, bf16_t* __restrict__ dst, int n) {
    int i = blockIdx.x * blockDim.x + threadIdx.x;
    int stride = gridDim.x * blockDim.x;
    for (; i < n; i += stride) dst[i] = (bf16_t)src[i];
}

// ---------------------------------------------------------------------------
// TDM descriptor helpers: build D# group0/group1 for a 2-D bf16 tile.
//   group0: count=1 | lds_addr | global_addr | type=2
//   group1: data_size=2B, tensor_dim0/1, tile_dim0/1, tensor_dim0_stride
// ---------------------------------------------------------------------------
__device__ __forceinline__ v4u tdm_group0(const void* gaddr, unsigned lds_byte_off) {
    unsigned long long ga = (unsigned long long)(uintptr_t)gaddr;
    v4u d = {};
    d[0] = 1u;                                          // count=1, user mode
    d[1] = lds_byte_off;                                // lds_addr [63:32]
    d[2] = (unsigned)(ga & 0xFFFFFFFFu);                // global_addr [95:64]
    d[3] = (unsigned)((ga >> 32) & 0x01FFFFFFu) | (2u << 30); // addr hi | type=2
    return d;
}
__device__ __forceinline__ v8i tdm_group1(unsigned tdim0, unsigned tdim1,
                                          unsigned tildim0, unsigned tildim1,
                                          unsigned stride0) {
    v8i d = {};
    d[0] = (int)(1u << 16);                             // data_size = 1 -> 2 bytes
    d[1] = (int)((tdim0 & 0xFFFFu) << 16);              // tensor_dim0 [79:48] lo
    d[2] = (int)(((tdim0 >> 16) & 0xFFFFu) | ((tdim1 & 0xFFFFu) << 16));
    d[3] = (int)(((tdim1 >> 16) & 0xFFFFu) | ((tildim0 & 0xFFFFu) << 16)); // tile_dim0
    d[4] = (int)(tildim1 & 0xFFFFu);                    // tile_dim1 (tile_dim2=0)
    d[5] = (int)stride0;                                // tensor_dim0_stride lo32
    return d;
}

// ---------------------------------------------------------------------------
// Stage 1: head-wise gate GEMM, TDM-staged through LDS + v_wmma bf16
// Grid: x = (BSROWS/64)*4 N-chunks, y = head, z = gate. Block 256 = 8 waves.
// Block tile: M=64, N=64. Waves: 4 (M) x 2 (N-half); wave = 16x32 (2 acc tiles).
// ---------------------------------------------------------------------------
__global__ void __launch_bounds__(256)
gate_gemm_bf16(const bf16_t* __restrict__ Xb,   // [BSROWS, DD]
               const bf16_t* __restrict__ Wb,   // [NGATE, HH, HDIM, HDIM]
               bf16_t* __restrict__ G)          // [NGATE, BSROWS, DD]
{
    __shared__ __align__(16) bf16_t lds[64 * HDIM + 64 * HDIM]; // X tile | W tile

    const int mt    = blockIdx.x >> 2;          // 0..511 (64-row tiles)
    const int nBase = (blockIdx.x & 3) * 64;    // 0,64,128,192 within head
    const int h     = blockIdx.y;
    const int gate  = blockIdx.z;
    const int wave  = threadIdx.x >> 5;         // 0..7
    const int lane  = threadIdx.x & 31;
    const int mSub  = wave >> 1;                // 0..3
    const int nHalf = wave & 1;                 // 0..1

    const int laneRow = lane & 15;
    const int laneHi  = lane >> 4;

    // TDM: move X tile (64 x 256) and W tile (64 x 256) into LDS.
    if (wave == 0) {
        const bf16_t* gX = Xb + (size_t)(mt * 64) * DD + h * HDIM;
        const bf16_t* gW = Wb + ((size_t)(gate * HH + h) * HDIM + nBase) * HDIM;
        v4u x0 = tdm_group0(gX, 0u);
        v8i x1 = tdm_group1(/*tensor*/ HDIM, 64, /*tile*/ HDIM, 64, /*stride*/ DD);
        v4u w0 = tdm_group0(gW, 64u * HDIM * 2u);
        v8i w1 = tdm_group1(HDIM, 64, HDIM, 64, HDIM);
        v4i z4 = {};
        v8i z8 = {};
        __builtin_amdgcn_tensor_load_to_lds(x0, x1, z4, z4, z8, 0);
        __builtin_amdgcn_tensor_load_to_lds(w0, w1, z4, z4, z8, 0);
        __builtin_amdgcn_s_wait_tensorcnt(0);
    }
    __syncthreads();

    // The TDM engine wrote lds[] behind the compiler's back. Escape the array
    // into an opaque asm with a memory clobber (so loads of it cannot fold to
    // undef), and perturb the index chain with an opaque zero offset — while
    // keeping the address chain rooted at the __shared__ array so the loads
    // still lower to native ds_load_b128 (not flat).
    unsigned ldsOff = 0;
    asm volatile("" : "+v"(ldsOff) : "v"((const void*)lds) : "memory");

    // A frag (16-bit 16x32): lanes 0-15 row=lane, K kb+{0..7,16..23};
    //                        lanes 16-31 row=lane-16, K kb+{8..15,24..31}
    const bf16_t* aLds = lds + ldsOff + (size_t)(mSub * 16 + laneRow) * HDIM + laneHi * 8;
    // B frag (16-bit 32x16): col N = lane&15; lanes 0-15 K=kb+{0..7,16..23}, 16-31 K=kb+{8..15,24..31}
    const bf16_t* wLds = lds + ldsOff + 64 * HDIM;

    v8f acc[2] = {};
    for (int kb = 0; kb < HDIM; kb += 32) {
        Frag128x2 fa;
        fa.u[0] = *(const uint4*)(aLds + kb);
        fa.u[1] = *(const uint4*)(aLds + kb + 16);
#pragma unroll
        for (int nt = 0; nt < 2; ++nt) {
            const bf16_t* wr = wLds + (size_t)(nHalf * 32 + nt * 16 + laneRow) * HDIM
                                    + laneHi * 8 + kb;
            Frag128x2 fb;
            fb.u[0] = *(const uint4*)(wr);
            fb.u[1] = *(const uint4*)(wr + 16);
            acc[nt] = __builtin_amdgcn_wmma_f32_16x16x32_bf16(
                false, fa.v, false, fb.v, (short)0, acc[nt], false, false);
        }
    }

    // C/D layout: VGPR r -> lanes 0-15 (M=r, N=lane), lanes 16-31 (M=8+r, N=lane-16)
#pragma unroll
    for (int nt = 0; nt < 2; ++nt) {
        const int col = h * HDIM + nBase + nHalf * 32 + nt * 16 + laneRow;
#pragma unroll
        for (int r = 0; r < 8; ++r) {
            const int row = mt * 64 + mSub * 16 + laneHi * 8 + r;
            G[((size_t)gate * BSROWS + row) * DD + col] = (bf16_t)acc[nt][r];
        }
    }
}

// ---------------------------------------------------------------------------
// Stage 2: chunked parallel scan (3 passes).  Recurrence per (b,d):
//   m_t = max(f_t + m, i_t);  c = e^{f+m-m'}c + e^{i-m'}tanh(z);  n likewise.
// Chunk summary from identity state (c=n=0, m=-1e30) + P = sum(f).
// ---------------------------------------------------------------------------
__global__ void __launch_bounds__(256)
scan_pass1(const bf16_t* __restrict__ G, float* __restrict__ cP, float* __restrict__ cM,
           float* __restrict__ cC, float* __restrict__ cN)
{
    const int tid = blockIdx.x * blockDim.x + threadIdx.x;   // BD*NC threads
    const int k   = tid / BD;
    const int idx = tid - k * BD;
    const int b = idx / DD, d = idx % DD;
    const size_t base = ((size_t)b * SS + (size_t)k * LC) * DD + d;
    const bf16_t* gz = G + 0 * (size_t)BSROWS * DD + base;
    const bf16_t* gi = G + 1 * (size_t)BSROWS * DD + base;
    const bf16_t* gf = G + 2 * (size_t)BSROWS * DD + base;

    float c = 0.f, n = 0.f, m = -1e30f, P = 0.f;
    for (int t = 0; t < LC; ++t) {
        const float zt = (float)gz[(size_t)t * DD];
        const float it = (float)gi[(size_t)t * DD];
        const float ft = (float)gf[(size_t)t * DD];
        P += ft;
        const float mn = fmaxf(ft + m, it);
        const float ih = __expf(it - mn);
        const float fh = __expf(ft + m - mn);
        c = fh * c + ih * tanhf(zt);
        n = fh * n + ih;
        m = mn;
    }
    cP[tid] = P; cM[tid] = m; cC[tid] = c; cN[tid] = n;
}

__global__ void __launch_bounds__(256)
scan_pass2(const float* __restrict__ cP, const float* __restrict__ cM,
           const float* __restrict__ cC, const float* __restrict__ cN,
           float* __restrict__ sC, float* __restrict__ sN, float* __restrict__ sM)
{
    const int idx = blockIdx.x * blockDim.x + threadIdx.x;   // BD threads
    if (idx >= BD) return;
    float c = 0.f, n = 0.f, m = 0.f;                         // reference initial state
    for (int k = 0; k < NC; ++k) {
        const size_t o = (size_t)k * BD + idx;
        sC[o] = c; sN[o] = n; sM[o] = m;                     // state entering chunk k
        const float P = cP[o], M = cM[o], Bc = cC[o], Bn = cN[o];
        const float mn = fmaxf(m + P, M);
        const float eh = __expf(m + P - mn);
        const float eb = __expf(M - mn);
        c = eh * c + eb * Bc;
        n = eh * n + eb * Bn;
        m = mn;
    }
}

__global__ void __launch_bounds__(256)
scan_pass3(const bf16_t* __restrict__ G,
           const float* __restrict__ sC, const float* __restrict__ sN,
           const float* __restrict__ sM, float* __restrict__ out)
{
    const int tid = blockIdx.x * blockDim.x + threadIdx.x;
    const int k   = tid / BD;
    const int idx = tid - k * BD;
    const int b = idx / DD, d = idx % DD;
    const size_t base = ((size_t)b * SS + (size_t)k * LC) * DD + d;
    const bf16_t* gz = G + 0 * (size_t)BSROWS * DD + base;
    const bf16_t* gi = G + 1 * (size_t)BSROWS * DD + base;
    const bf16_t* gf = G + 2 * (size_t)BSROWS * DD + base;
    const bf16_t* go = G + 3 * (size_t)BSROWS * DD + base;
    float* hp = out + base;

    float c = sC[tid], n = sN[tid], m = sM[tid];
    for (int t = 0; t < LC; ++t) {
        const float zt = (float)gz[(size_t)t * DD];
        const float it = (float)gi[(size_t)t * DD];
        const float ft = (float)gf[(size_t)t * DD];
        const float ot = (float)go[(size_t)t * DD];
        const float o  = 1.0f / (1.0f + __expf(-ot));
        const float mn = fmaxf(ft + m, it);
        const float ih = __expf(it - mn);
        const float fh = __expf(ft + m - mn);
        c = fh * c + ih * tanhf(zt);
        n = fh * n + ih;
        m = mn;
        hp[(size_t)t * DD] = o * (c / (n + 1e-8f));
    }
}

// ---------------------------------------------------------------------------
// Launch
// ---------------------------------------------------------------------------
extern "C" void kernel_launch(void* const* d_in, const int* in_sizes, int n_in,
                              void* d_out, int out_size, void* d_ws, size_t ws_size,
                              hipStream_t stream) {
    const float* x = (const float*)d_in[0];
    const float* Wg[NGATE] = { (const float*)d_in[1],   // Wz
                               (const float*)d_in[2],   // Wi
                               (const float*)d_in[3],   // Wf
                               (const float*)d_in[4] }; // Wo
    float* out = (float*)d_out;

    // Workspace layout
    char* ws = (char*)d_ws;
    const size_t offWb = (size_t)BSROWS * DD * 2;                       // after Xb
    const size_t offG  = offWb + (size_t)NGATE * HH * HDIM * HDIM * 2;  // after Wb
    const size_t offCk = offG + (size_t)NGATE * BSROWS * DD * 2;        // after G
    bf16_t* Xb = (bf16_t*)ws;
    bf16_t* Wb = (bf16_t*)(ws + offWb);
    bf16_t* G  = (bf16_t*)(ws + offG);
    float*  cP = (float*)(ws + offCk);
    float*  cM = cP + (size_t)BD * NC;
    float*  cC = cM + (size_t)BD * NC;
    float*  cN = cC + (size_t)BD * NC;
    float*  sC = cN + (size_t)BD * NC;
    float*  sN = sC + (size_t)BD * NC;
    float*  sM = sN + (size_t)BD * NC;

    // Stage 0: conversions
    const int nx = BSROWS * DD;
    cvt_f32_to_bf16<<<2048, 256, 0, stream>>>(x, Xb, nx);
    const int nw = HH * HDIM * HDIM;
    for (int g = 0; g < NGATE; ++g)
        cvt_f32_to_bf16<<<(nw + 255) / 256, 256, 0, stream>>>(Wg[g], Wb + (size_t)g * nw, nw);

    // Stage 1: TDM + WMMA gate GEMM
    dim3 grid((BSROWS / 64) * 4, HH, NGATE);   // (2048, 4, 4)
    gate_gemm_bf16<<<grid, 256, 0, stream>>>(Xb, Wb, G);

    // Stage 2: chunked scan
    scan_pass1<<<(BD * NC) / 256, 256, 0, stream>>>(G, cP, cM, cC, cN);
    scan_pass2<<<(BD + 255) / 256, 256, 0, stream>>>(cP, cM, cC, cN, sC, sN, sM);
    scan_pass3<<<(BD * NC) / 256, 256, 0, stream>>>(G, sC, sN, sM, out);
}
